// SelectiveSSM_40235253629560
// MI455X (gfx1250) — matrix operand order repaired and checked by reading, not compiled, not run on hardware
//
#include <hip/hip_runtime.h>
#include <hip/hip_bf16.h>

// ---------- types ----------
typedef __attribute__((ext_vector_type(16))) __bf16   v16bf;
typedef __attribute__((ext_vector_type(4)))  __bf16   v4bf;
typedef __attribute__((ext_vector_type(8)))  float    v8f;
typedef __attribute__((ext_vector_type(4)))  unsigned v4u;
typedef __attribute__((ext_vector_type(8)))  unsigned v8u;

// f32 -> bf16 round-to-nearest-even (used only in one-shot pack/epilogue kernels,
// never in GEMM inner loops)
__device__ __forceinline__ __bf16 f2bf(float f) {
  unsigned u = __builtin_bit_cast(unsigned, f);
  u += 0x7FFFu + ((u >> 16) & 1u);
  unsigned short h = (unsigned short)(u >> 16);
  return __builtin_bit_cast(__bf16, h);
}
__device__ __forceinline__ float bf2f(__bf16 b) {
  unsigned u = (unsigned)__builtin_bit_cast(unsigned short, b) << 16;
  return __builtin_bit_cast(float, u);
}

// A fragment (16x32 MxK, bf16): lane0-15 -> M=lane, K 0..15;
// lane16-31 -> K 16..31.  Two 16B loads.
__device__ __forceinline__ v16bf ldA(const __bf16* __restrict__ p) {
  v4u lo = *(const v4u*)p;          // 8 bf16
  v4u hi = *(const v4u*)(p + 8);    // 8 bf16
  v8u u;
  u[0] = lo[0]; u[1] = lo[1]; u[2] = lo[2]; u[3] = lo[3];
  u[4] = hi[0]; u[5] = hi[1]; u[6] = hi[2]; u[7] = hi[3];
  return __builtin_bit_cast(v16bf, u);
}
// B fragment (32x16 KxN, bf16): lane = column, 16 contiguous K per lane (32B).
__device__ __forceinline__ v16bf ldB(const __bf16* __restrict__ p) {
  return __builtin_bit_cast(v16bf, *(const v8u*)p);
}

// C = A(MxK bf16, lda) * W(NxK bf16, ldb)^T, f32 C (ldc).
// 256 thr = 8 waves as 4(M) x 2(N); wave tile 32 x (NF*16); block 128 x (NF*32).
// Inner loop: (2 + NF) fragment loads -> 2*NF WMMAs, no conversion VALU.
template <int NF>
__global__ __launch_bounds__(256)
void gemm_bf16(const __bf16* __restrict__ A, int lda,
               const __bf16* __restrict__ Bw, int ldb,
               float* __restrict__ C, int ldc, int K) {
  const int lane  = threadIdx.x & 31;
  const int wave  = threadIdx.x >> 5;
  const int wm    = wave & 3, wn = wave >> 2;
  const int mBase = blockIdx.y * 128 + wm * 32;
  const int nBase = blockIdx.x * (NF * 32) + wn * (NF * 16);
  const int l15   = lane & 15;
  const int aklo  = (lane & 16) ? 16 : 0;
  const int bklo  = (lane & 16) ? 16 : 0;

  const __bf16* aRow0 = A + (long)(mBase + l15) * lda + aklo;
  const __bf16* aRow1 = aRow0 + (long)16 * lda;
  const __bf16* bRow  = Bw + (long)(nBase + l15) * ldb + bklo;

  v8f acc[2][NF];
  #pragma unroll
  for (int mi = 0; mi < 2; ++mi)
    #pragma unroll
    for (int nf = 0; nf < NF; ++nf) acc[mi][nf] = v8f{};

  for (int k0 = 0; k0 < K; k0 += 32) {
    v16bf a0 = ldA(aRow0 + k0);
    v16bf a1 = ldA(aRow1 + k0);
    #pragma unroll
    for (int nf = 0; nf < NF; ++nf) {
      v16bf b = ldB(bRow + (long)nf * 16 * ldb + k0);
      acc[0][nf] = __builtin_amdgcn_wmma_f32_16x16x32_bf16(false, a0, false, b, (short)0, acc[0][nf], false, false);
      acc[1][nf] = __builtin_amdgcn_wmma_f32_16x16x32_bf16(false, a1, false, b, (short)0, acc[1][nf], false, false);
    }
  }

  // C/D layout: VGPR r, lanes 0-15 -> M=r, lanes 16-31 -> M=8+r; N = lane&15.
  const int rsel = (lane & 16) ? 8 : 0;
  #pragma unroll
  for (int mi = 0; mi < 2; ++mi) {
    float* cp = C + (long)(mBase + mi * 16 + rsel) * ldc + nBase + l15;
    #pragma unroll
    for (int nf = 0; nf < NF; ++nf)
      #pragma unroll
      for (int r = 0; r < 8; ++r) cp[(long)r * ldc + nf * 16] = acc[mi][nf][r];
  }
}

// One-shot f32 -> bf16 pack (4 elements/thread)
__global__ __launch_bounds__(256)
void pack_bf16(const float* __restrict__ src, __bf16* __restrict__ dst) {
  long i = (long)blockIdx.x * 256 + threadIdx.x;
  float4 f = ((const float4*)src)[i];
  v4bf o;
  o[0] = f2bf(f.x); o[1] = f2bf(f.y); o[2] = f2bf(f.z); o[3] = f2bf(f.w);
  ((v4bf*)dst)[i] = o;
}

// Zero-pad W_x (33x2048) to (64x2048) bf16 so the ssm projection is a WMMA GEMM.
__global__ __launch_bounds__(256)
void pad_wx_bf16(const float* __restrict__ Wx, __bf16* __restrict__ Wxp) {
  int idx = blockIdx.x * 256 + threadIdx.x;            // over 64*2048
  float v = ((idx >> 11) < 33) ? Wx[idx] : 0.f;
  Wxp[idx] = f2bf(v);
}

// Depthwise causal conv (width 4) + bias + SiLU; writes bf16 x_conv panel.
__global__ __launch_bounds__(256)
void conv_silu_kernel(const float* __restrict__ xz, const float* __restrict__ cw,
                      const float* __restrict__ cb, __bf16* __restrict__ xcb) {
  long idx = (long)blockIdx.x * 256 + threadIdx.x;     // over B*T*2048
  int  d  = (int)(idx & 2047);
  long bt = idx >> 11;
  int  t  = (int)(bt & 2047);
  float4 w = *(const float4*)(cw + (long)d * 4);       // conv_w[d,0,0..3]
  const float* base = xz + bt * 4096 + d;
  float acc = cb[d];
  if (t >= 3) acc += w.x * base[-3 * 4096];
  if (t >= 2) acc += w.y * base[-2 * 4096];
  if (t >= 1) acc += w.z * base[-1 * 4096];
  acc += w.w * base[0];
  float s = acc / (1.f + __expf(-acc));                // silu
  xcb[idx] = f2bf(s);
}

// Selective scan: thread <-> (b,d); 16 f32 states in registers; T-serial.
// ssm row (per b,t): [0:16]=B_t, [16:32]=C_t, [32]=dt_raw.
// B_t/C_t: one lane-indexed load + __shfl broadcast across the wave.
__global__ __launch_bounds__(256)
void scan_kernel(const float* __restrict__ ssm, const float* __restrict__ xz,
                 const __bf16* __restrict__ xcb, __bf16* __restrict__ yb,
                 const float* __restrict__ dt_w, const float* __restrict__ dt_b,
                 const float* __restrict__ A_log, const float* __restrict__ Dp) {
  const int b    = blockIdx.x >> 3;
  const int d    = ((blockIdx.x & 7) << 8) + threadIdx.x;
  const int lane = threadIdx.x & 31;
  const float dtw = dt_w[d], dtb = dt_b[d], Dpar = Dp[d];
  float as[16], h[16];
  #pragma unroll
  for (int s = 0; s < 16; ++s) { as[s] = -__expf(A_log[s]); h[s] = 0.f; }
  const long btBase = (long)b * 2048;
  for (int t = 0; t < 2048; ++t) {
    const long bt = btBase + t;
    float bc  = ssm[bt * 64 + lane];                   // lanes 0-15: B_s; 16-31: C_s
    float dtr = ssm[bt * 64 + 32];
    float xv  = bf2f(xcb[bt * 2048 + d]);
    float zv  = xz[bt * 4096 + 2048 + d];
    float sp  = (dtr > 20.f) ? dtr : log1pf(__expf(dtr));  // softplus
    float dt  = sp * dtw + dtb;
    float dtx = dt * xv;
    float y = 0.f;
    #pragma unroll
    for (int s = 0; s < 16; ++s) {
      float Bs = __shfl(bc, s, 32);
      float Cs = __shfl(bc, s + 16, 32);
      float dA = __expf(as[s] * dt);
      h[s] = dA * h[s] + dtx * Bs;
      y += h[s] * Cs;
    }
    y += Dpar * xv;
    float sz = zv / (1.f + __expf(-zv));               // silu(z)
    yb[bt * 2048 + d] = f2bf(y * sz);
  }
}

extern "C" void kernel_launch(void* const* d_in, const int* in_sizes, int n_in,
                              void* d_out, int out_size, void* d_ws, size_t ws_size,
                              hipStream_t stream) {
  (void)in_sizes; (void)n_in; (void)out_size; (void)ws_size;
  const float* x     = (const float*)d_in[0];   // (4,2048,1024)
  const float* W_in  = (const float*)d_in[1];   // (4096,1024)
  const float* cw    = (const float*)d_in[2];   // (2048,1,4)
  const float* cb    = (const float*)d_in[3];   // (2048,)
  const float* W_x   = (const float*)d_in[4];   // (33,2048)
  const float* dtw   = (const float*)d_in[5];
  const float* dtb   = (const float*)d_in[6];
  const float* A_log = (const float*)d_in[7];
  const float* Dp    = (const float*)d_in[8];
  const float* W_out = (const float*)d_in[9];   // (1024,2048)
  float* out = (float*)d_out;                   // (4,2048,1024) f32

  // ---- workspace layout (bytes), total ~233 MB ----
  char* w = (char*)d_ws;
  float*  xz    = (float*)w;                     w += (size_t)8192 * 4096 * 4; // x_br | z (f32)
  float*  ssm   = (float*)w;                     w += (size_t)8192 * 64 * 4;   // B|C|dt (f32)
  __bf16* xb    = (__bf16*)w;                    w += (size_t)8192 * 1024 * 2; // x bf16
  __bf16* Winb  = (__bf16*)w;                    w += (size_t)4096 * 1024 * 2;
  __bf16* Woutb = (__bf16*)w;                    w += (size_t)1024 * 2048 * 2;
  __bf16* Wxpb  = (__bf16*)w;                    w += (size_t)64 * 2048 * 2;
  __bf16* xcb   = (__bf16*)w;                    w += (size_t)8192 * 2048 * 2; // x_conv bf16
  __bf16* yb    = (__bf16*)w;                    // y bf16

  dim3 blk(256);
  // 0) one-shot bf16 packs (amortized: each element converted exactly once)
  pack_bf16<<<dim3((8192 * 1024) / 1024), blk, 0, stream>>>(x, xb);
  pack_bf16<<<dim3((4096 * 1024) / 1024), blk, 0, stream>>>(W_in, Winb);
  pack_bf16<<<dim3((1024 * 2048) / 1024), blk, 0, stream>>>(W_out, Woutb);
  pad_wx_bf16<<<dim3((64 * 2048) / 256), blk, 0, stream>>>(W_x, Wxpb);
  // 1) xz = x @ W_in^T        M=8192 N=4096 K=1024
  gemm_bf16<4><<<dim3(4096 / 128, 8192 / 128), blk, 0, stream>>>(xb, 1024, Winb, 1024, xz, 4096, 1024);
  // 2) x_conv = silu(depthwise_conv(x_br) + b)  -> bf16 panel
  conv_silu_kernel<<<dim3((8192 * 2048) / 256), blk, 0, stream>>>(xz, cw, cb, xcb);
  // 3) ssm = x_conv @ Wxp^T   M=8192 N=64 K=2048
  gemm_bf16<2><<<dim3(1, 8192 / 128), blk, 0, stream>>>(xcb, 2048, Wxpb, 2048, ssm, 64, 2048);
  // 4) selective scan -> y bf16 panel
  scan_kernel<<<dim3(32), blk, 0, stream>>>(ssm, xz, xcb, yb, dtw, dtb, A_log, Dp);
  // 5) out = y @ W_out^T      M=8192 N=1024 K=2048
  gemm_bf16<4><<<dim3(1024 / 128, 8192 / 128), blk, 0, stream>>>(yb, 2048, Woutb, 2048, out, 1024, 2048);
}
